// VectorQuantizer_78443282694565
// MI455X (gfx1250) — matrix operand order repaired and checked by reading, not compiled, not run on hardware
//
#include <hip/hip_runtime.h>

#define DIMS 64
#define KCODES 1024
#define ROWS_PER_WG 128
#define WAVES_PER_WG 8
#define CB_PAD 72   // bf16 elems per codebook row in LDS (64 + 8 pad -> 144B row stride)
#define CN_PAD 68   // floats per transposed-norm row (64 + 4 pad -> conflict-free b128)

typedef __attribute__((ext_vector_type(16))) __bf16 v16bf;
typedef __attribute__((ext_vector_type(8)))  float  v8f;
typedef __attribute__((ext_vector_type(4)))  float  nf4;   // native vec for nontemporal builtins
typedef __attribute__((ext_vector_type(4)))  int    vi4;   // native vec for async-LDS builtin

union V16BF { v16bf v; unsigned short u[16]; };
union BFLD  { v16bf v; uint4 q[2]; };
union F4CVT { float4 h; nf4 n; };

__device__ __forceinline__ unsigned short f32_to_bf16_rne(float f) {
    union { float f; unsigned int u; } c; c.f = f;
    unsigned int u = c.u;
    u += 0x7FFFu + ((u >> 16) & 1u);   // round-to-nearest-even
    return (unsigned short)(u >> 16);
}

__device__ __forceinline__ void pack8(V16BF& dst, int off, float4 a, float4 b) {
    dst.u[off + 0] = f32_to_bf16_rne(a.x); dst.u[off + 1] = f32_to_bf16_rne(a.y);
    dst.u[off + 2] = f32_to_bf16_rne(a.z); dst.u[off + 3] = f32_to_bf16_rne(a.w);
    dst.u[off + 4] = f32_to_bf16_rne(b.x); dst.u[off + 5] = f32_to_bf16_rne(b.y);
    dst.u[off + 6] = f32_to_bf16_rne(b.z); dst.u[off + 7] = f32_to_bf16_rne(b.w);
}

// Pass 1: per-code squared norms + bf16 codebook copy.
__global__ void vq_prep(const float* __restrict__ cb,
                        unsigned short* __restrict__ cb_bf,
                        float* __restrict__ cnorm) {
    int k = blockIdx.x * blockDim.x + threadIdx.x;   // one thread per code
    const float* row = cb + (size_t)k * DIMS;
    float s = 0.f;
#pragma unroll 8
    for (int j = 0; j < DIMS; ++j) {
        float v = row[j];
        s += v * v;
        cb_bf[(size_t)k * DIMS + j] = f32_to_bf16_rne(v);
    }
    cnorm[k] = s;
}

// Pass 2: WMMA distance GEMM + argmin + gather + quantized output + loss partials.
__global__ __launch_bounds__(256) void vq_main(
    const float* __restrict__ z, const float* __restrict__ cb,
    const unsigned short* __restrict__ cb_bf, const float* __restrict__ cnorm,
    float* __restrict__ out, float* __restrict__ partials) {

    extern __shared__ char smem[];
    unsigned short* cb_lds = (unsigned short*)smem;                     // 1024*72*2 = 147456 B
    float* cn_lds  = (float*)(smem + (size_t)KCODES * CB_PAD * 2);      // 16*68*4 = 4352 B
    int*   idx_lds = (int*)((char*)cn_lds + 16 * CN_PAD * 4);           // 512 B
    float* red_lds = (float*)((char*)idx_lds + ROWS_PER_WG * 4);       // 32 B

    const int tid  = threadIdx.x;
    const int wave = tid >> 5;
    const int lane = tid & 31;
    const int half = lane >> 4;     // lane group (ISA 16-bit A/B layouts)
    const int n    = lane & 15;

    // ---- hoist this wave's z rows (global stream) so they overlap LDS staging ----
    // A 16x32 operand, this code's contiguous k order: lane (half,m=n): u[0..15]=K(32s+16h+0..15)
    const size_t arow = (size_t)blockIdx.x * ROWS_PER_WG + wave * 16 + n;
    const float4* z4 = (const float4*)(z + arow * DIMS);
    float4 zc0 = z4[4 * half + 0],  zc1 = z4[4 * half + 1];
    float4 zc2 = z4[4 * half + 2],  zc3 = z4[4 * half + 3];
    float4 zc4 = z4[8 + 4 * half],  zc5 = z4[9 + 4 * half];
    float4 zc6 = z4[10 + 4 * half], zc7 = z4[11 + 4 * half];

    // ---- stage bf16 codebook (row-padded) into LDS ----
#if defined(__has_builtin) && __has_builtin(__builtin_amdgcn_global_load_async_to_lds_b128)
    for (int row = tid; row < KCODES; row += 256) {
        const uint4* src = (const uint4*)(cb_bf + (size_t)row * DIMS);   // 128 B/row
        uint4* dst = (uint4*)(cb_lds + (size_t)row * CB_PAD);            // 144 B stride
#pragma unroll
        for (int j = 0; j < 8; ++j)
            __builtin_amdgcn_global_load_async_to_lds_b128(
                (vi4*)(src + j), (vi4*)(dst + j), 0, 0);
    }
#else
    for (int row = tid; row < KCODES; row += 256) {
        const uint4* src = (const uint4*)(cb_bf + (size_t)row * DIMS);
        uint4* dst = (uint4*)(cb_lds + (size_t)row * CB_PAD);
#pragma unroll
        for (int j = 0; j < 8; ++j) dst[j] = src[j];
    }
#endif
    // ---- stage norms transposed: cn_lds[n*CN_PAD + t] = cnorm[t*16+n] ----
    for (int i = tid; i < KCODES; i += 256)
        cn_lds[(i & 15) * CN_PAD + (i >> 4)] = cnorm[i];

    // ---- build A fragments while staging is in flight ----
    V16BF a0, a1;
    pack8(a0, 0, zc0, zc1);   // K = 16h .. 16h+7
    pack8(a0, 8, zc2, zc3);   // K = 16h+8 ..
    pack8(a1, 0, zc4, zc5);   // K = 32+16h ..
    pack8(a1, 8, zc6, zc7);   // K = 40+16h ..

#if defined(__has_builtin) && __has_builtin(__builtin_amdgcn_global_load_async_to_lds_b128)
    asm volatile("s_wait_asynccnt 0x0" ::: "memory");
#endif
    __syncthreads();

    float bestD[8]; int bestI[8];
#pragma unroll
    for (int r = 0; r < 8; ++r) { bestD[r] = 3.4e38f; bestI[r] = 0; }

    // ---- sweep 64 code tiles of 16; dist = ||e||^2 - 2 z.e (||z||^2 constant per row) ----
    const float* cnrow = cn_lds + n * CN_PAD;   // this lane's 64 norms, contiguous
    for (int t4 = 0; t4 < KCODES / 64; ++t4) {
        float4 cn4 = *(const float4*)(cnrow + t4 * 4);   // norms for tiles 4*t4..+3
#pragma unroll
        for (int u = 0; u < 4; ++u) {
            const int t = t4 * 4 + u;
            // B 32x16 operand, same contiguous k order as A: lane (half,col=n): u[0..15] = K(32s + 16h + 0..15)
            const unsigned short* brow = cb_lds + (size_t)(t * 16 + n) * CB_PAD + 16 * half;
            BFLD b0, b1;
            const uint4* bq0 = (const uint4*)brow;          // 16B aligned (144*r + 32*half)
            const uint4* bq1 = (const uint4*)(brow + 32);   // +64 B
            b0.q[0] = bq0[0]; b0.q[1] = bq0[1];
            b1.q[0] = bq1[0]; b1.q[1] = bq1[1];

            v8f acc = {};
            acc = __builtin_amdgcn_wmma_f32_16x16x32_bf16(false, a0.v, false, b0.v,
                                                          (short)0, acc, false, false);
            acc = __builtin_amdgcn_wmma_f32_16x16x32_bf16(false, a1.v, false, b1.v,
                                                          (short)0, acc, false, false);
            const float cn = (&cn4.x)[u];
            const int code = t * 16 + n;
#pragma unroll
            for (int r = 0; r < 8; ++r) {                    // element (M=r+8*half, N=n)
                float d = fmaf(-2.0f, acc[r], cn);
                if (d < bestD[r]) { bestD[r] = d; bestI[r] = code; }
            }
        }
    }

    // ---- cross-lane argmin over the 16 columns (first-index tie-break) ----
#pragma unroll
    for (int r = 0; r < 8; ++r) {
        float d = bestD[r]; int i = bestI[r];
#pragma unroll
        for (int off = 1; off < 16; off <<= 1) {         // stays inside half-groups
            float od = __shfl_xor(d, off, 32);
            int   oi = __shfl_xor(i, off, 32);
            if (od < d || (od == d && oi < i)) { d = od; i = oi; }
        }
        if (n == 0) idx_lds[wave * 16 + r + 8 * half] = i;
    }
    __syncthreads();

    // ---- gather fp32 codebook rows, emit quantized (== straight-through fwd), loss partial ----
    const int m  = lane >> 1;
    const int c0 = (lane & 1) * 32;
    const size_t grow = (size_t)blockIdx.x * ROWS_PER_WG + wave * 16 + m;
    const int qi = idx_lds[wave * 16 + m];
    const float4* crow = (const float4*)(cb + (size_t)qi * DIMS + c0);
    const float4* zrow = (const float4*)(z + grow * DIMS + c0);
    nf4* orow = (nf4*)(out + grow * DIMS + c0);
    float sq = 0.f;
#pragma unroll
    for (int j = 0; j < 8; ++j) {
        float4 q  = crow[j];
        float4 zv = zrow[j];
        float dx = zv.x - q.x, dy = zv.y - q.y, dz = zv.z - q.z, dw = zv.w - q.w;
        sq += dx * dx + dy * dy + dz * dz + dw * dw;
        F4CVT cvt; cvt.h = q;
        __builtin_nontemporal_store(cvt.n, orow + j);
    }
#pragma unroll
    for (int off = 16; off; off >>= 1) sq += __shfl_xor(sq, off, 32);
    if (lane == 0) red_lds[wave] = sq;
    __syncthreads();
    if (tid == 0) {
        float s = 0.f;
#pragma unroll
        for (int w = 0; w < WAVES_PER_WG; ++w) s += red_lds[w];
        partials[blockIdx.x] = s;   // unconditional write: no zero-init, deterministic
    }
}

// Pass 3: ordered reduction of per-WG partials -> scalar loss.
__global__ void vq_finalize(const float* __restrict__ partials, int nb,
                            float scale, float* __restrict__ loss_out) {
    float s = 0.f;
    for (int i = 0; i < nb; ++i) s += partials[i];
    *loss_out = s * scale;   // (0.25 + 1.0) * mean
}

extern "C" void kernel_launch(void* const* d_in, const int* in_sizes, int n_in,
                              void* d_out, int out_size, void* d_ws, size_t ws_size,
                              hipStream_t stream) {
    const float* z  = (const float*)d_in[0];
    const float* cb = (const float*)d_in[1];
    const int nvec = in_sizes[0] / DIMS;     // 65536
    const int kc   = in_sizes[1] / DIMS;     // 1024

    char* ws = (char*)d_ws;
    float* partials        = (float*)ws;                               // <= 4 KB
    unsigned short* cb_bf  = (unsigned short*)(ws + 4096);             // 128 KB
    float* cnorm           = (float*)(ws + 4096 + (size_t)kc * DIMS * 2);

    float* out  = (float*)d_out;
    float* loss = out + (out_size - 1);

    vq_prep<<<kc / 256, 256, 0, stream>>>(cb, cb_bf, cnorm);

    const int nblocks = nvec / ROWS_PER_WG;  // 512
    const size_t lds_bytes = (size_t)KCODES * CB_PAD * 2   // bf16 codebook, padded
                           + (size_t)16 * CN_PAD * 4       // transposed norms
                           + ROWS_PER_WG * 4               // argmin indices
                           + WAVES_PER_WG * 4;             // loss reduce
    vq_main<<<nblocks, 256, lds_bytes, stream>>>(z, cb, cb_bf, cnorm, out, partials);

    vq_finalize<<<1, 1, 0, stream>>>(partials, nblocks,
                                     1.25f / (float)in_sizes[0], loss);
}
